// NeuralDevice_79010218377372
// MI455X (gfx1250) — matrix operand order repaired and checked by reading, not compiled, not run on hardware
//
#include <hip/hip_runtime.h>
#include <hip/hip_bf16.h>

typedef __attribute__((ext_vector_type(16))) __bf16 v16bf;
typedef __attribute__((ext_vector_type(8)))  __bf16 v8bf;
typedef __attribute__((ext_vector_type(8)))  float  v8f;

// ---------------------------------------------------------------------------
// A-fragment builder: 16-bit A matrix 16x32 layout (CDNA5 ISA 7.12.2):
//   lane l<16 : row=l,    elements 0..7 -> K=kbase+0..7,  8..15 -> K=kbase+16..23
//   lane l>=16: row=l-16, elements 0..7 -> K=kbase+8..15, 8..15 -> K=kbase+24..31
// lhi = (lane/16)*8. Source rows are f32; converted to bf16 with optional scale
// (used to fold the segment-mean 1/count into the GEMM input).
// ---------------------------------------------------------------------------
__device__ inline v16bf make_afrag(const float* __restrict__ rp, int kbase,
                                   int lhi, float scale) {
  const float4* p0 = (const float4*)(rp + kbase + lhi);
  const float4* p1 = (const float4*)(rp + kbase + lhi + 16);
  float4 f0 = p0[0], f1 = p0[1];
  float4 f2 = p1[0], f3 = p1[1];
  v16bf a;
  a[0]  = (__bf16)(f0.x * scale); a[1]  = (__bf16)(f0.y * scale);
  a[2]  = (__bf16)(f0.z * scale); a[3]  = (__bf16)(f0.w * scale);
  a[4]  = (__bf16)(f1.x * scale); a[5]  = (__bf16)(f1.y * scale);
  a[6]  = (__bf16)(f1.z * scale); a[7]  = (__bf16)(f1.w * scale);
  a[8]  = (__bf16)(f2.x * scale); a[9]  = (__bf16)(f2.y * scale);
  a[10] = (__bf16)(f2.z * scale); a[11] = (__bf16)(f2.w * scale);
  a[12] = (__bf16)(f3.x * scale); a[13] = (__bf16)(f3.y * scale);
  a[14] = (__bf16)(f3.z * scale); a[15] = (__bf16)(f3.w * scale);
  return a;
}

// B-fragment: 32x16 bf16, lanes 0-15 hold K 0..7,16..23 of column n=lane,
// lanes 16-31 hold K 8..15,24..31 of column n=lane-16. Weights are staged in LDS
// transposed (WT[n][k], contiguous in k) so this is 2x ds_load_b128.
__device__ inline v16bf load_bfrag(const __bf16* __restrict__ p) {
  v8bf lo = *(const v8bf*)p;
  v8bf hi = *(const v8bf*)(p + 16);
  v16bf r;
#pragma unroll
  for (int i = 0; i < 8; ++i) { r[i] = lo[i]; r[i + 8] = hi[i]; }
  return r;
}

// ---------------------------------------------------------------------------
// Utility kernels
// ---------------------------------------------------------------------------
__global__ __launch_bounds__(256) void zero_kernel(float4* __restrict__ p,
                                                   long long n4) {
  long long i = (long long)blockIdx.x * blockDim.x + threadIdx.x;
  long long stride = (long long)gridDim.x * blockDim.x;
  float4 z; z.x = 0.f; z.y = 0.f; z.z = 0.f; z.w = 0.f;
  for (; i < n4; i += stride) p[i] = z;
}

__global__ __launch_bounds__(256) void count_kernel(const int* __restrict__ idx,
                                                    float* __restrict__ cnt,
                                                    int n) {
  int i = blockIdx.x * blockDim.x + threadIdx.x;
  if (i < n) unsafeAtomicAdd(&cnt[(unsigned)idx[i]], 1.0f);
}

// ---------------------------------------------------------------------------
// Stage 1: per-node h = relu(x @ Wr + br), scatter-add into s[idx] ([B,128]).
// One wave handles a 16-node tile: 8 N-tiles x 2 K-steps of
// v_wmma_f32_16x16x32_bf16, then 64 no-return global_atomic_add_f32 per lane
// into the L2-resident accumulator (64MB << 192MB L2). All per-lane address
// math is unsigned 32-bit so the compiler can use SADDR + voffset (GVS)
// addressing instead of 64-bit VALU address arithmetic.
// ---------------------------------------------------------------------------
__global__ __launch_bounds__(256) void eye_scatter_kernel(
    const float* __restrict__ x,    // [N,64]
    const int*   __restrict__ idx,  // [N]
    const float* __restrict__ Wr,   // [64,128]
    const float* __restrict__ br,   // [128]
    float*       __restrict__ s) {  // [B,128] accumulators
  __shared__ __bf16 WT[128 * 64];   // Wr transposed, bf16: WT[n][k]

  // Stage weights (tiny, L2-resident source).
  for (int i = threadIdx.x; i < 128 * 64; i += 256) {
    int n = i >> 6, k = i & 63;
    WT[i] = (__bf16)Wr[k * 128 + n];
  }
  __syncthreads();

  const int lane = threadIdx.x & 31;
  const int wid  = threadIdx.x >> 5;
  const int lmod = lane & 15;
  const int lhi  = (lane >> 4) << 3;   // 0 or 8

  const int tile = blockIdx.x * 8 + wid;
  const int base = tile * 16;
  const float* rp = x + (unsigned)(base + lmod) * 64u;

  // A fragments for K=0..31 and K=32..63 (kept in registers for all 8 N-tiles)
  v16bf A0 = make_afrag(rp, 0, lhi, 1.0f);
  v16bf A1 = make_afrag(rp, 32, lhi, 1.0f);

  // Segment ids for the 8 rows this lane's D elements belong to.
  const int4* ip = (const int4*)(idx + base + lhi);
  int4 s01 = ip[0], s23 = ip[1];
  int seg[8] = {s01.x, s01.y, s01.z, s01.w, s23.x, s23.y, s23.z, s23.w};

#pragma unroll
  for (int nt = 0; nt < 8; ++nt) {
    const int col = nt * 16 + lmod;
    const __bf16* wp = WT + col * 64 + lhi;  // kbase=(lane/16)*8
    v16bf B0 = load_bfrag(wp);
    v16bf B1 = load_bfrag(wp + 32);
    v8f acc = {};
    acc = __builtin_amdgcn_wmma_f32_16x16x32_bf16(false, A0, false, B0,
                                                  (short)0, acc, false, false);
    acc = __builtin_amdgcn_wmma_f32_16x16x32_bf16(false, A1, false, B1,
                                                  (short)0, acc, false, false);
    const float bv = br[col];
#pragma unroll
    for (int v = 0; v < 8; ++v) {
      float val = fmaxf(acc[v] + bv, 0.0f);
      // 32-bit element offset: B*128 elements = 8.4M < 2^31
      unsigned off = ((unsigned)seg[v] << 7) | (unsigned)col;
      unsafeAtomicAdd(&s[off], val);
    }
  }
}

// ---------------------------------------------------------------------------
// Generic dense row-tile GEMM: Out[r, ocol0+0..127] = act(In[r,:K] @ W[:, wcol0+0..127] + b)
// Optional per-row 1/count scale (segment mean) folded into A conversion.
// One wave per 16-row tile; W slice staged transposed+bf16 in LDS.
// ---------------------------------------------------------------------------
template <int KIN>
__global__ __launch_bounds__(256) void gemm_rows_kernel(
    const float* __restrict__ In,    // [R, KIN] f32
    const float* __restrict__ W,     // [KIN, nfull] f32
    const float* __restrict__ bias,  // [nfull]
    const float* __restrict__ cnt,   // [R] or nullptr
    float*       __restrict__ Out,   // [R, ostride]
    int ostride, int ocol0, int wcol0, int nfull, int relu) {
  extern __shared__ __bf16 WT[];     // [128][KIN] transposed slice

  for (int i = threadIdx.x; i < 128 * KIN; i += 256) {
    int n = i / KIN, k = i % KIN;
    WT[i] = (__bf16)W[k * nfull + wcol0 + n];
  }
  __syncthreads();

  const int lane = threadIdx.x & 31;
  const int wid  = threadIdx.x >> 5;
  const int lmod = lane & 15;
  const int lhi  = (lane >> 4) << 3;

  const int tile = blockIdx.x * 8 + wid;
  const int base = tile * 16;
  const int row  = base + lmod;
  const float* rp = In + (unsigned)row * (unsigned)KIN;
  const float scale = cnt ? (1.0f / fmaxf(cnt[(unsigned)row], 1.0f)) : 1.0f;

  v16bf A[KIN / 32];
#pragma unroll
  for (int kf = 0; kf < KIN / 32; ++kf)
    A[kf] = make_afrag(rp, kf * 32, lhi, scale);

#pragma unroll
  for (int nt = 0; nt < 8; ++nt) {
    const int col = nt * 16 + lmod;
    v8f acc = {};
#pragma unroll
    for (int kf = 0; kf < KIN / 32; ++kf) {
      v16bf B = load_bfrag(WT + col * KIN + kf * 32 + lhi);
      acc = __builtin_amdgcn_wmma_f32_16x16x32_bf16(false, A[kf], false, B,
                                                    (short)0, acc, false, false);
    }
    const float bv = bias[wcol0 + col];
#pragma unroll
    for (int v = 0; v < 8; ++v) {
      float val = acc[v] + bv;
      if (relu) val = fmaxf(val, 0.0f);
      unsigned off = (unsigned)(base + v + lhi) * (unsigned)ostride +
                     (unsigned)(ocol0 + col);
      Out[off] = val;
    }
  }
}

// ---------------------------------------------------------------------------
// Launch
// ---------------------------------------------------------------------------
extern "C" void kernel_launch(void* const* d_in, const int* in_sizes, int n_in,
                              void* d_out, int out_size, void* d_ws, size_t ws_size,
                              hipStream_t stream) {
  const float* x0  = (const float*)d_in[0];
  const float* x1  = (const float*)d_in[1];
  const int*   i0  = (const int*)d_in[2];
  const int*   i1  = (const int*)d_in[3];
  const float* Wr0 = (const float*)d_in[4];
  const float* br0 = (const float*)d_in[5];
  const float* Wc0 = (const float*)d_in[6];
  const float* bc0 = (const float*)d_in[7];
  const float* Wr1 = (const float*)d_in[8];
  const float* br1 = (const float*)d_in[9];
  const float* Wc1 = (const float*)d_in[10];
  const float* bc1 = (const float*)d_in[11];
  const float* Wb1 = (const float*)d_in[12];
  const float* bb1 = (const float*)d_in[13];
  const float* Wb2 = (const float*)d_in[14];
  const float* bb2 = (const float*)d_in[15];
  float* out = (float*)d_out;

  const int N = in_sizes[2];       // 1048576
  const int B = out_size / 128;    // 65536

  // Workspace layout (f32): s0[B*128] s1[B*128] cnt0[B] cnt1[B] z[B*256] h[B*256]
  float* s0   = (float*)d_ws;
  float* s1   = s0 + (long long)B * 128;
  float* cnt0 = s1 + (long long)B * 128;
  float* cnt1 = cnt0 + B;
  float* z    = cnt1 + B;
  float* h    = z + (long long)B * 256;
  (void)ws_size; (void)n_in;

  // 1. zero accumulators + counts (z/h/out fully overwritten later)
  long long n4 = ((long long)2 * B * 128 + 2 * B) / 4;
  zero_kernel<<<4096, 256, 0, stream>>>((float4*)d_ws, n4);

  // 2. segment counts
  count_kernel<<<N / 256, 256, 0, stream>>>(i0, cnt0, N);
  count_kernel<<<N / 256, 256, 0, stream>>>(i1, cnt1, N);

  // 3. per-node GEMM + relu + scatter-add (memory-bound stage)
  eye_scatter_kernel<<<N / 128, 256, 0, stream>>>(x0, i0, Wr0, br0, s0);
  eye_scatter_kernel<<<N / 128, 256, 0, stream>>>(x1, i1, Wr1, br1, s1);

  // 4. per-eye: e = relu(mean @ Wc + bc) -> z[:, eye*128 .. ]
  gemm_rows_kernel<128><<<B / 128, 256, 128 * 128 * 2, stream>>>(
      s0, Wc0, bc0, cnt0, z, 256, 0, 0, 128, 1);
  gemm_rows_kernel<128><<<B / 128, 256, 128 * 128 * 2, stream>>>(
      s1, Wc1, bc1, cnt1, z, 256, 128, 0, 128, 1);

  // 5. h = relu(z @ Wb1 + bb1), two 128-column slices (LDS slice = 64KB)
  gemm_rows_kernel<256><<<B / 128, 256, 128 * 256 * 2, stream>>>(
      z, Wb1, bb1, nullptr, h, 256, 0, 0, 256, 1);
  gemm_rows_kernel<256><<<B / 128, 256, 128 * 256 * 2, stream>>>(
      z, Wb1, bb1, nullptr, h, 256, 128, 128, 256, 1);

  // 6. out = h @ Wb2 + bb2
  gemm_rows_kernel<256><<<B / 128, 256, 128 * 256 * 2, stream>>>(
      h, Wb2, bb2, nullptr, out, 128, 0, 0, 128, 0);
}
